// GCN_77644418777840
// MI455X (gfx1250) — matrix operand order repaired and checked by reading, not compiled, not run on hardware
//
#include <hip/hip_runtime.h>
#include <hip/hip_bf16.h>

// ---------------------------------------------------------------------------
// GCN forward, bf16 feature path: 4x gcn_conv, WMMA bf16 GEMMs (gfx1250).
//   N_NODES=100000, N_EDGES=400000, EMB_DIM=HID=512
// Features/hw kept as bf16 (halves gather traffic, hw fits in 192MB L2);
// aggregation accumulates in f32 via global_atomic_add_f32.
// ---------------------------------------------------------------------------

#define HID 512
#define NT_PER_BLOCK 4     // 4 N-tiles of 16 -> 64 cols per block
#define MT_PER_WAVE  2     // 2 M-tiles of 16 -> 32 rows per wave
#define KTILES 16          // 512 / 32

typedef unsigned int  uint;
typedef unsigned short ushort;
typedef __attribute__((ext_vector_type(16))) __bf16 v16bf;
typedef __attribute__((ext_vector_type(8)))  float  v8f;

__device__ __forceinline__ ushort f2bf(float f) {
    union { float f; uint u; } x; x.f = f;
    uint u = x.u + 0x7FFFu + ((x.u >> 16) & 1u);   // round-to-nearest-even
    return (ushort)(u >> 16);
}
__device__ __forceinline__ float bf2f(ushort b) {
    union { uint u; float f; } x; x.u = ((uint)b) << 16;
    return x.f;
}

// ---------------- degree / norm ----------------
__global__ void k_init_deg(float* deg, int n) {
    int i = blockIdx.x * blockDim.x + threadIdx.x;
    if (i < n) deg[i] = 1.0f;                           // self-loop
}
__global__ void k_deg_accum(const int* __restrict__ dst, float* deg, int e) {
    int i = blockIdx.x * blockDim.x + threadIdx.x;
    if (i < e) atomicAdd(&deg[dst[i]], 1.0f);
}
__global__ void k_dinv(float* deg, int n) {
    int i = blockIdx.x * blockDim.x + threadIdx.x;
    if (i < n) deg[i] = rsqrtf(deg[i]);                 // in place: deg -> dinv
}

// ---------------- embedding gather -> bf16 features ----------------
__global__ void k_embed(const int* __restrict__ x, const float* __restrict__ emb,
                        uint4* __restrict__ h, long total8) {      // 8 bf16 per uint4
    long i = (long)blockIdx.x * blockDim.x + threadIdx.x;
    if (i >= total8) return;
    long node = i >> 6;                 // 64 uint4 per row
    int  c8   = (int)(i & 63);
    const float4* e4 = (const float4*)emb + (long)x[node] * 128 + c8 * 2;
    float4 a = e4[0], b = e4[1];
    union { uint4 q; ushort u[8]; } o;
    o.u[0]=f2bf(a.x); o.u[1]=f2bf(a.y); o.u[2]=f2bf(a.z); o.u[3]=f2bf(a.w);
    o.u[4]=f2bf(b.x); o.u[5]=f2bf(b.y); o.u[6]=f2bf(b.z); o.u[7]=f2bf(b.w);
    h[i] = o.q;
}

// ---------------- pack W[512,512] f32 -> bf16 B-fragment layout ----------
// packW[ntile][ktile][lane][e] : lane = {g=K-half, n}, e -> K = ktile*32+g*16+e
__global__ void k_pack_w(const float* __restrict__ W, ushort* __restrict__ pw) {
    int t = blockIdx.x * blockDim.x + threadIdx.x;      // 32*16*32 = 16384
    if (t >= 32 * KTILES * 32) return;
    int lane  = t & 31;
    int ktile = (t >> 5) & (KTILES - 1);
    int ntile = t >> 9;
    int n  = ntile * 16 + (lane & 15);
    int kb = ktile * 32 + (lane >> 4) * 16;
    ushort* d = pw + (((size_t)ntile * KTILES + ktile) * 32 + lane) * 16;
    #pragma unroll
    for (int e = 0; e < 16; ++e) d[e] = f2bf(W[(size_t)(kb + e) * HID + n]);
}

// ---------------- WMMA GEMM: hw_bf16 = h_bf16 @ W  -------------------------
// block: 256 thr (8 waves); wave computes 32 rows x 64 cols (2x4 WMMA tiles).
__global__ __launch_bounds__(256) void k_gemm_wmma(
        const ushort* __restrict__ A,          // bf16 row-major [nRows,512]
        const ushort* __restrict__ pw,         // packed bf16 B fragments
        ushort* __restrict__ out, int nRows) { // bf16 row-major [nRows,512]
    __shared__ alignas(16) ushort ldsB[NT_PER_BLOCK * KTILES * 32 * 16]; // 64 KB

    // stage this block's 512x64 B panel (contiguous in packed layout)
    const uint4* srcB = (const uint4*)(pw + (size_t)blockIdx.y * NT_PER_BLOCK * KTILES * 32 * 16);
    uint4* dstB = (uint4*)ldsB;
    for (int i = threadIdx.x; i < NT_PER_BLOCK * KTILES * 32 * 2; i += 256) dstB[i] = srcB[i];
    __syncthreads();

    const int wid  = threadIdx.x >> 5;
    const int lane = threadIdx.x & 31;
    const int m    = lane & 15;
    const int g    = lane >> 4;
    const long mt0 = ((long)blockIdx.x * 8 + wid) * MT_PER_WAVE;

    const uint4* arow[MT_PER_WAVE];
    #pragma unroll
    for (int t = 0; t < MT_PER_WAVE; ++t) {
        long row  = (mt0 + t) * 16 + m;
        long rowc = row < nRows ? row : (long)(nRows - 1);   // clamp edge loads
        arow[t] = (const uint4*)(A + rowc * HID);            // 64 uint4 per row
    }

    v8f acc[MT_PER_WAVE][NT_PER_BLOCK];
    #pragma unroll
    for (int t = 0; t < MT_PER_WAVE; ++t)
        #pragma unroll
        for (int nt = 0; nt < NT_PER_BLOCK; ++nt) acc[t][nt] = (v8f){};

    for (int kk = 0; kk < KTILES; ++kk) {
        // A fragment: lane(m,g): uint4s at (kb+g*16)/8 = kk*4+2*g and kk*4+2*g+1
        union { v16bf v; uint4 q[2]; } af[MT_PER_WAVE];
        #pragma unroll
        for (int t = 0; t < MT_PER_WAVE; ++t) {
            af[t].q[0] = arow[t][kk * 4 + 2 * g];
            af[t].q[1] = arow[t][kk * 4 + 2 * g + 1];
        }
        #pragma unroll
        for (int nt = 0; nt < NT_PER_BLOCK; ++nt) {
            const v16bf bfr = *(const v16bf*)&ldsB[((nt * KTILES + kk) * 32 + lane) * 16];
            #pragma unroll
            for (int t = 0; t < MT_PER_WAVE; ++t)
                acc[t][nt] = __builtin_amdgcn_wmma_f32_16x16x32_bf16(
                                 false, af[t].v, false, bfr, (short)0, acc[t][nt], false, false);
        }
    }

    // ---- store C (bf16): lane l, vgpr r -> row = r + 8*g, col = l&15 ----
    const int colbase = blockIdx.y * (NT_PER_BLOCK * 16) + (lane & 15);
    if ((mt0 + MT_PER_WAVE) * 16 <= nRows) {               // fast path: no guards
        #pragma unroll
        for (int t = 0; t < MT_PER_WAVE; ++t)
            #pragma unroll
            for (int nt = 0; nt < NT_PER_BLOCK; ++nt)
                #pragma unroll
                for (int r = 0; r < 8; ++r) {
                    long orow = (mt0 + t) * 16 + r + 8 * g;
                    out[orow * HID + colbase + nt * 16] = f2bf(acc[t][nt][r]);
                }
    } else {
        #pragma unroll
        for (int t = 0; t < MT_PER_WAVE; ++t)
            #pragma unroll
            for (int nt = 0; nt < NT_PER_BLOCK; ++nt)
                #pragma unroll
                for (int r = 0; r < 8; ++r) {
                    long orow = (mt0 + t) * 16 + r + 8 * g;
                    if (orow < nRows)
                        out[orow * HID + colbase + nt * 16] = f2bf(acc[t][nt][r]);
                }
    }
}

// ---------------- aggregation init: out_f32 = bias + dinv^2 * hw ----------
__global__ void k_agg_init(const uint4* __restrict__ hw, const float* __restrict__ dinv,
                           const float* __restrict__ bias, float* __restrict__ out, long total8) {
    long i = (long)blockIdx.x * blockDim.x + threadIdx.x;
    if (i >= total8) return;
    long node = i >> 6;
    int  c8   = (int)(i & 63);
    float di = dinv[node];
    float w  = di * di;
    union { uint4 q; ushort u[8]; } v; v.q = hw[i];
    const float4* b4 = (const float4*)bias + c8 * 2;
    float4 ba = b4[0], bb = b4[1];
    float4 o0 = { ba.x + bf2f(v.u[0]) * w, ba.y + bf2f(v.u[1]) * w,
                  ba.z + bf2f(v.u[2]) * w, ba.w + bf2f(v.u[3]) * w };
    float4 o1 = { bb.x + bf2f(v.u[4]) * w, bb.y + bf2f(v.u[5]) * w,
                  bb.z + bf2f(v.u[6]) * w, bb.w + bf2f(v.u[7]) * w };
    ((float4*)out)[i * 2]     = o0;
    ((float4*)out)[i * 2 + 1] = o1;
}

// ---------------- edge scatter: out[dst] += norm * hw[src]  (warp/edge) ---
__global__ void k_agg_edges(const ushort* __restrict__ hw, const int* __restrict__ srcI,
                            const int* __restrict__ dstI, const float* __restrict__ dinv,
                            float* __restrict__ out, int nEdges) {
    int e    = (blockIdx.x * blockDim.x + threadIdx.x) >> 5;
    int lane = threadIdx.x & 31;
    if (e >= nEdges) return;
    int s = srcI[e], d = dstI[e];
    float w = dinv[s] * dinv[d];
    const uint* hs = (const uint*)(hw + (long)s * HID);   // 256 uints (2 bf16 each)
    float*      od = out + (long)d * HID;
    #pragma unroll
    for (int k = 0; k < 8; ++k) {
        uint u = hs[lane + 32 * k];                       // coalesced 128B/wave
        int idx = (lane + 32 * k) * 2;
        atomicAdd(&od[idx],     bf2f((ushort)(u & 0xFFFFu)) * w);
        atomicAdd(&od[idx + 1], bf2f((ushort)(u >> 16)) * w);
    }
}

// ---------------- relu + f32 -> bf16 feature conversion -------------------
__global__ void k_relu_cvt(const float* __restrict__ in, uint4* __restrict__ out, long total8) {
    long i = (long)blockIdx.x * blockDim.x + threadIdx.x;
    if (i >= total8) return;
    float4 a = ((const float4*)in)[i * 2];
    float4 b = ((const float4*)in)[i * 2 + 1];
    union { uint4 q; ushort u[8]; } o;
    o.u[0]=f2bf(fmaxf(a.x,0.f)); o.u[1]=f2bf(fmaxf(a.y,0.f));
    o.u[2]=f2bf(fmaxf(a.z,0.f)); o.u[3]=f2bf(fmaxf(a.w,0.f));
    o.u[4]=f2bf(fmaxf(b.x,0.f)); o.u[5]=f2bf(fmaxf(b.y,0.f));
    o.u[6]=f2bf(fmaxf(b.z,0.f)); o.u[7]=f2bf(fmaxf(b.w,0.f));
    out[i] = o.q;
}

// ---------------- layer 4: hw4 = h_bf16 . W4  (warp per node) -------------
__global__ void k_matvec4(const ushort* __restrict__ h, const float* __restrict__ W4,
                          float* __restrict__ out, int n) {
    int node = (blockIdx.x * blockDim.x + threadIdx.x) >> 5;
    int lane = threadIdx.x & 31;
    if (node >= n) return;
    const uint* hr = (const uint*)(h + (long)node * HID);
    float s = 0.f;
    #pragma unroll
    for (int k = 0; k < 8; ++k) {
        uint u = hr[lane + 32 * k];
        int idx = (lane + 32 * k) * 2;
        s += bf2f((ushort)(u & 0xFFFFu)) * W4[idx];
        s += bf2f((ushort)(u >> 16))     * W4[idx + 1];
    }
    #pragma unroll
    for (int off = 16; off > 0; off >>= 1) s += __shfl_down(s, off, 32);
    if (lane == 0) out[node] = s;
}

__global__ void k_agg4_init(const float* __restrict__ hw4, const float* __restrict__ dinv,
                            const float* __restrict__ b4, float* __restrict__ s4, int n) {
    int i = blockIdx.x * blockDim.x + threadIdx.x;
    if (i < n) { float di = dinv[i]; s4[i] = b4[0] + di * di * hw4[i]; }
}
__global__ void k_agg4_edges(const float* __restrict__ hw4, const int* __restrict__ srcI,
                             const int* __restrict__ dstI, const float* __restrict__ dinv,
                             float* __restrict__ s4, int nEdges) {
    int e = blockIdx.x * blockDim.x + threadIdx.x;
    if (e >= nEdges) return;
    int s = srcI[e], d = dstI[e];
    atomicAdd(&s4[d], dinv[s] * dinv[d] * hw4[s]);
}
__global__ void k_sigmoid(const float* __restrict__ s4, float* __restrict__ out, int n) {
    int i = blockIdx.x * blockDim.x + threadIdx.x;
    if (i < n) out[i] = 1.0f / (1.0f + expf(-s4[i]));
}

// ---------------------------------------------------------------------------
extern "C" void kernel_launch(void* const* d_in, const int* in_sizes, int n_in,
                              void* d_out, int out_size, void* d_ws, size_t ws_size,
                              hipStream_t stream) {
    const int*   x    = (const int*)d_in[0];
    const int*   eidx = (const int*)d_in[1];
    const float* emb  = (const float*)d_in[2];
    const float* W1   = (const float*)d_in[3];
    const float* b1   = (const float*)d_in[4];
    const float* W2   = (const float*)d_in[5];
    const float* b2   = (const float*)d_in[6];
    const float* W3   = (const float*)d_in[7];
    const float* b3   = (const float*)d_in[8];
    const float* W4   = (const float*)d_in[9];
    const float* b4   = (const float*)d_in[10];
    float* out = (float*)d_out;

    const int N = in_sizes[0];            // 100000
    const int E = in_sizes[1] / 2;        // 400000
    const int* srcI = eidx;
    const int* dstI = eidx + E;

    // workspace layout
    char* ws = (char*)d_ws;
    float*  dinv  = (float*)(ws);                                  // N f32
    float*  s4    = (float*)(ws + (size_t)512 * 1024);             // N f32
    ushort* packW = (ushort*)(ws + (size_t)1024 * 1024);           // 512KB
    float*  hw4   = (float*)(ws + (size_t)1536 * 1024 + 65536);    // N f32
    ushort* bufX  = (ushort*)(ws + (size_t)2 * 1024 * 1024);       // N*512 bf16
    ushort* bufY  = bufX + (size_t)N * HID;                        // N*512 bf16
    float*  bufF  = (float*)(bufY + (size_t)N * HID);              // N*512 f32

    const int  TB  = 256;
    const long T8  = (long)N * (HID / 8);                 // uint4 (8-elem) count
    const int  gN  = (N + TB - 1) / TB;
    const int  gE  = (E + TB - 1) / TB;
    const int  g8  = (int)((T8 + TB - 1) / TB);
    const int  gEw = (int)(((long)E * 32 + TB - 1) / TB); // warp per edge
    const int  gNw = (int)(((long)N * 32 + TB - 1) / TB); // warp per node
    const int  gPk = (32 * KTILES * 32 + TB - 1) / TB;
    const int  mtiles = (N + 15) / 16;
    dim3 gemmGrid((mtiles + 8 * MT_PER_WAVE - 1) / (8 * MT_PER_WAVE),
                  HID / (NT_PER_BLOCK * 16), 1);

    // ---- norm: deg -> dinv ----
    k_init_deg <<<gN, TB, 0, stream>>>(dinv, N);
    k_deg_accum<<<gE, TB, 0, stream>>>(dstI, dinv, E);
    k_dinv     <<<gN, TB, 0, stream>>>(dinv, N);

    // ---- h0 = bf16(emb[x]) ----
    k_embed<<<g8, TB, 0, stream>>>(x, emb, (uint4*)bufX, T8);

    // ---- layers 1..3: gemm -> agg_init -> agg_edges -> relu_cvt ----
    const float* Ws[3] = { W1, W2, W3 };
    const float* bs[3] = { b1, b2, b3 };
    for (int l = 0; l < 3; ++l) {
        k_pack_w   <<<gPk, TB, 0, stream>>>(Ws[l], packW);
        k_gemm_wmma<<<gemmGrid, TB, 0, stream>>>(bufX, packW, bufY, N);
        k_agg_init <<<g8, TB, 0, stream>>>((const uint4*)bufY, dinv, bs[l], bufF, T8);
        k_agg_edges<<<gEw, TB, 0, stream>>>(bufY, srcI, dstI, dinv, bufF, E);
        k_relu_cvt <<<g8, TB, 0, stream>>>(bufF, (uint4*)bufX, T8);
    }

    // ---- layer 4: dot with W4, aggregate, sigmoid ----
    k_matvec4   <<<gNw, TB, 0, stream>>>(bufX, W4, hw4, N);
    k_agg4_init <<<gN, TB, 0, stream>>>(hw4, dinv, b4, s4, N);
    k_agg4_edges<<<gE, TB, 0, stream>>>(hw4, srcI, dstI, dinv, s4, E);
    k_sigmoid   <<<gN, TB, 0, stream>>>(s4, out, N);

    (void)n_in; (void)out_size; (void)ws_size;
}
